// ContRepDecoder_61512521614075
// MI455X (gfx1250) — matrix operand order repaired and checked by reading, not compiled, not run on hardware
//
#include <hip/hip_runtime.h>
#include <hip/hip_bf16.h>
#include <math.h>

typedef __attribute__((ext_vector_type(16))) _Float16 v16h;
typedef __attribute__((ext_vector_type(8)))  float    v8f;

#define C_CH    128
#define D_DIM   176
#define NT_MLP  11
#define KK_MLP  6
#define N_LAYER 6
#define OUT_CH  45
#define OUT_PAD 48
#define NT_POST 3
#define KK_POST 4
#define XC      16
#define EPSF    1e-7f
#define PIF     3.14159265358979f

// padded LDS row strides (halves), both 16B-aligned, bank-conflict-friendly
#define HST 200   // hbuf stride  (>=192)
#define CST 136   // ctxbuf stride (>=128)

// workspace layout (in _Float16 units)
#define WSW_HALVES  (N_LAYER*KK_MLP*NT_MLP*512)   // 202752 halves : swizzled MLP W
#define WSPW_HALVES (KK_POST*NT_POST*512)          // 6144 halves  : swizzled post W

__device__ inline void atomicMinF(float* a, float v) {
  if (v >= 0.f) atomicMin((int*)a, __float_as_int(v));
  else          atomicMax((unsigned int*)a, __float_as_uint(v));
}

__global__ void k_init_mn(float* __restrict__ mn) {
  int t = threadIdx.x;
  if (t < 48) mn[t] = __builtin_inff();
}

// Pre-swizzle weights (f32 -> f16) into per-lane WMMA B-fragment layout:
// fragment = 32 lanes x 16 halves; lane l holds column nt*16+(l&15),
// K values 32*kk + (l>=16?16:0) + 2v{+1}.  K padded 176 -> 192 with zeros.
__global__ void k_swizzle(const float* __restrict__ mlp_w,
                          const float* __restrict__ post_w,
                          _Float16* __restrict__ wsw,
                          _Float16* __restrict__ wspw) {
  int tid = blockIdx.x * 256 + threadIdx.x;
  const int MLPD = N_LAYER * KK_MLP * NT_MLP * 256;   // half2 slots
  if (tid < MLPD) {
    int d = tid & 255;
    int f = tid >> 8;
    int nt = f % NT_MLP; f /= NT_MLP;
    int kk = f % KK_MLP; f /= KK_MLP;
    int L  = f;                                       // 0..5 = blk*3+layer
    int lane = d >> 3, v = d & 7;
    int k   = 32*kk + ((lane >= 16) ? 16 : 0) + 2*v;
    int col = nt*16 + (lane & 15);
    float w0 = (k     < D_DIM) ? mlp_w[(L*D_DIM + k    )*D_DIM + col] : 0.f;
    float w1 = (k + 1 < D_DIM) ? mlp_w[(L*D_DIM + k + 1)*D_DIM + col] : 0.f;
    _Float16* o = wsw + ((L*KK_MLP*NT_MLP + kk*NT_MLP + nt)*512 + lane*16 + v*2);
    o[0] = (_Float16)w0; o[1] = (_Float16)w1;
  } else {
    int t2 = tid - MLPD;
    if (t2 < KK_POST * NT_POST * 256) {
      int d = t2 & 255;
      int f = t2 >> 8;
      int nt = f % NT_POST;
      int kk = f / NT_POST;
      int lane = d >> 3, v = d & 7;
      int k   = 32*kk + ((lane >= 16) ? 16 : 0) + 2*v;   // always < 128
      int col = nt*16 + (lane & 15);
      float w0 = (col < OUT_CH) ? post_w[k*OUT_CH + col]       : 0.f;
      float w1 = (col < OUT_CH) ? post_w[(k + 1)*OUT_CH + col] : 0.f;
      _Float16* o = wspw + ((kk*NT_POST + nt)*512 + lane*16 + v*2);
      o[0] = (_Float16)w0; o[1] = (_Float16)w1;
    }
  }
}

// Global min over all query points of rel, per (batch, corner, component).
__global__ void k_minred(const float* __restrict__ ext,
                         const float* __restrict__ qcoord,
                         float* __restrict__ mn) {
  __shared__ float sm[3];
  int tid = threadIdx.x;
  if (tid < 3) sm[tid] = __builtin_inff();
  __syncthreads();
  int g   = blockIdx.x * 256 + tid;          // 0..65535 query point
  int b   = g >> 15;
  int lin = g & 32767;
  int r   = blockIdx.y;                       // corner 0..7
  int ci = r >> 2, cj = (r >> 1) & 1, ck = r & 1;
  float qc[3], org[3], vox[3];
  int id[3];
  for (int c = 0; c < 3; ++c) {
    const float* e = ext + ((b*3 + c) << 12);
    org[c] = e[0];
    vox[c] = fabsf(e[(1 << 8) + (1 << 4) + 1] - org[c]);
    qc[c]  = qcoord[((b*3 + c) << 15) + lin];
    int v = (int)floorf((qc[c] - org[c]) / vox[c] + 0.5f);
    id[c] = min(max(v, 0), XC - 2);
  }
  int cell = ((id[0] + ci) << 8) | ((id[1] + cj) << 4) | (id[2] + ck);
  for (int c = 0; c < 3; ++c) {
    float ccv = ext[((b*3 + c) << 12) + cell];
    float rel = fmaxf(ccv - qc[c], -0.5f*vox[c] + EPSF);
    atomicMinF(&sm[c], rel);
  }
  __syncthreads();
  if (tid < 3) atomicMinF(&mn[b*24 + r*3 + tid], sm[tid]);
}

union FragH { v16h v; float4 f4[2]; };

__global__ void __launch_bounds__(128) k_main(
    const float* __restrict__ cv,
    const float* __restrict__ ext,
    const float* __restrict__ qvsg,
    const float* __restrict__ qcoord,
    const float* __restrict__ mlp_b,
    const float* __restrict__ post_b,
    const _Float16* __restrict__ wsw,
    const _Float16* __restrict__ wspw,
    const float* __restrict__ mn,
    float* __restrict__ out) {
  __shared__ _Float16 hbuf[128][HST];     // [corner*16+p][K 0..191]
  __shared__ _Float16 ctxbuf[128][CST];   // residual ctx (128 ch)
  __shared__ float predbuf[8][16][OUT_PAD];
  __shared__ float qcs[16][3];
  __shared__ int   idxs[16][3];
  __shared__ float tls[16][3];
  __shared__ float wgt[128];
  __shared__ float voxs[3], orgs[3], qvss[3];

  const int tid  = threadIdx.x;
  const int lane = tid & 31;
  const int wave = tid >> 5;
  const int g0   = blockIdx.x * 16;       // 16 consecutive query points
  const int b    = g0 >> 15;
  const int lin0 = g0 & 32767;

  // ---- phase 0: per-batch constants
  if (tid < 3) {
    const float* e = ext + ((b*3 + tid) << 12);
    float o = e[0];
    orgs[tid] = o;
    voxs[tid] = fabsf(e[(1 << 8) + (1 << 4) + 1] - o);
    qvss[tid] = qvsg[b*3 + tid];
  }
  __syncthreads();
  // ---- per-point coords + base lattice index
  if (tid < 16) {
    int lin = lin0 + tid;
    for (int c = 0; c < 3; ++c) {
      float qc = qcoord[((b*3 + c) << 15) + lin];
      qcs[tid][c] = qc;
      int v = (int)floorf((qc - orgs[c]) / voxs[c] + 0.5f);
      idxs[tid][c] = min(max(v, 0), XC - 2);
    }
  }
  __syncthreads();

  // ---- phase 1: one thread per row = (corner, point): features + cv gather
  {
    const int r = tid >> 4, p = tid & 15;
    const int ci = r >> 2, cj = (r >> 1) & 1, ck = r & 1;
    const int ix = idxs[p][0] + ci, iy = idxs[p][1] + cj, iz = idxs[p][2] + ck;
    const int cell = (ix << 8) | (iy << 4) | iz;
    float feats[48];
    float reln[3];
    for (int c = 0; c < 3; ++c) {
      float ccv = ext[((b*3 + c) << 12) + cell];
      float rel = fmaxf(ccv - qcs[p][c], -0.5f*voxs[c] + EPSF);
      float m   = mn[b*24 + r*3 + c];
      float rn  = (rel - m) / (1.5f * voxs[c]);
      reln[c] = rn;
      feats[c]     = qvss[c];
      feats[3 + c] = ccv;
      feats[6 + c] = qcs[p][c];
      feats[9 + c] = rn;
      float base = 2.f * PIF * rn;      // SIGMA = 1
      float fr = 1.f;
      for (int f = 0; f < 6; ++f) {
        float ang = base * fr;
        feats[12 + c*12 + f]     = __sinf(ang);
        feats[12 + c*12 + 6 + f] = __cosf(ang);
        fr *= 2.f;
      }
    }
    if (r == 0) {   // trilinear t from corner (0,0,0)
      for (int c = 0; c < 3; ++c) {
        float gc = fminf(fmaxf((reln[c] - 0.5f) * 2.f, -1.f + EPSF), 1.f - EPSF);
        tls[p][c] = (gc + 1.f) * 0.5f;
      }
    }
    for (int c = 0; c < 48; ++c) hbuf[tid][C_CH + c] = (_Float16)feats[c];
    for (int c = D_DIM; c < 192; ++c) hbuf[tid][c] = (_Float16)0.f;   // K pad
    const float* cvp = cv + (((long)b * C_CH) << 12) + cell;
    for (int c = 0; c < C_CH; ++c) {
      _Float16 h = (_Float16)cvp[(long)c << 12];
      hbuf[tid][c]   = h;
      ctxbuf[tid][c] = h;
    }
  }
  __syncthreads();
  // ---- phase 2: trilinear weight per row (i<->t[2], k<->t[0] as in reference)
  {
    const int r = tid >> 4, p = tid & 15;
    const int ci = r >> 2, cj = (r >> 1) & 1, ck = r & 1;
    float t0 = tls[p][0], t1 = tls[p][1], t2 = tls[p][2];
    wgt[tid] = (ci ? t2 : 1.f - t2) * (cj ? t1 : 1.f - t1) * (ck ? t0 : 1.f - t0);
  }
  __syncthreads();

  // ---- WMMA phase: each wave owns M-tiles {wave, wave+4}; rows disjoint
  for (int tile = wave; tile < 8; tile += 4) {
    const int row0 = tile * 16;
    const int mloc = (lane >> 4) << 3;           // 0 or 8  (C/D layout)
    const int ncol = lane & 15;
    const int arow = row0 + (lane & 15);         // A layout: lanes 0-15 = M rows
    // ---- 2 blocks x 3 layers of D=176 SiLU MLP
    for (int blk = 0; blk < 2; ++blk) {
      for (int l = 0; l < 3; ++l) {
        const int L = blk*3 + l;
        v8f acc[NT_MLP];
        for (int nt = 0; nt < NT_MLP; ++nt) {
          float bv = mlp_b[L*D_DIM + nt*16 + ncol];
          v8f a;
          for (int q = 0; q < 8; ++q) a[q] = bv;
          acc[nt] = a;
        }
        for (int kk = 0; kk < KK_MLP; ++kk) {
          FragH af;
          const int kb = 32*kk + ((lane >> 4) << 4); // A: lanes>=16 take K+16
          af.f4[0] = *reinterpret_cast<const float4*>(&hbuf[arow][kb]);
          af.f4[1] = *reinterpret_cast<const float4*>(&hbuf[arow][kb + 8]);
          for (int nt = 0; nt < NT_MLP; ++nt) {
            FragH bf;
            const _Float16* bp = wsw + ((L*KK_MLP + kk)*NT_MLP + nt)*512 + lane*16;
            bf.f4[0] = reinterpret_cast<const float4*>(bp)[0];
            bf.f4[1] = reinterpret_cast<const float4*>(bp)[1];
            acc[nt] = __builtin_amdgcn_wmma_f32_16x16x32_f16(
                false, af.v, false, bf.v, (short)0, acc[nt], false, false);
          }
        }
        // SiLU + writeback (all A reads of this layer already consumed)
        for (int nt = 0; nt < NT_MLP; ++nt) {
          for (int q = 0; q < 8; ++q) {
            float x = acc[nt][q];
            float s = x / (1.f + __expf(-x));
            hbuf[row0 + mloc + q][nt*16 + ncol] = (_Float16)s;
          }
        }
      }
      // residual: ctx += h[:, :128]; h[:, :128] = ctx  (wave-private rows)
      for (int e = lane; e < 16 * C_CH; e += 32) {
        int rr = e >> 7, cc2 = e & 127;
        float nv = (float)ctxbuf[row0 + rr][cc2] + (float)hbuf[row0 + rr][cc2];
        _Float16 nh = (_Float16)nv;
        ctxbuf[row0 + rr][cc2] = nh;
        hbuf[row0 + rr][cc2]   = nh;
      }
    }
    // ---- post projection 128 -> 48(45)
    v8f pacc[NT_POST];
    for (int nt = 0; nt < NT_POST; ++nt) {
      int col = nt*16 + ncol;
      float bv = (col < OUT_CH) ? post_b[col] : 0.f;
      v8f a;
      for (int q = 0; q < 8; ++q) a[q] = bv;
      pacc[nt] = a;
    }
    for (int kk = 0; kk < KK_POST; ++kk) {
      FragH af;
      const int kb = 32*kk + ((lane >> 4) << 4);
      af.f4[0] = *reinterpret_cast<const float4*>(&ctxbuf[arow][kb]);
      af.f4[1] = *reinterpret_cast<const float4*>(&ctxbuf[arow][kb + 8]);
      for (int nt = 0; nt < NT_POST; ++nt) {
        FragH bf;
        const _Float16* bp = wspw + (kk*NT_POST + nt)*512 + lane*16;
        bf.f4[0] = reinterpret_cast<const float4*>(bp)[0];
        bf.f4[1] = reinterpret_cast<const float4*>(bp)[1];
        pacc[nt] = __builtin_amdgcn_wmma_f32_16x16x32_f16(
            false, af.v, false, bf.v, (short)0, pacc[nt], false, false);
      }
    }
    // trilinear-weighted per-corner prediction (deterministic, no atomics)
    for (int nt = 0; nt < NT_POST; ++nt) {
      for (int q = 0; q < 8; ++q) {
        int m = mloc + q;
        predbuf[tile][m][nt*16 + ncol] = pacc[nt][q] * wgt[row0 + m];
      }
    }
  }
  __syncthreads();
  // ---- blend 8 corners, write (b, OUT, x, y, z)
  for (int e = tid; e < OUT_CH * 16; e += 128) {
    int o = e >> 4, p = e & 15;
    float s = 0.f;
    for (int t = 0; t < 8; ++t) s += predbuf[t][p][o];
    out[(((long)b * OUT_CH + o) << 15) + lin0 + p] = s;
  }
}

extern "C" void kernel_launch(void* const* d_in, const int* in_sizes, int n_in,
                              void* d_out, int out_size, void* d_ws, size_t ws_size,
                              hipStream_t stream) {
  (void)in_sizes; (void)n_in; (void)out_size; (void)ws_size;
  const float* cv  = (const float*)d_in[0];   // context_v        (2,128,16,16,16)
  const float* ext = (const float*)d_in[1];   // spatial extent   (2,3,16,16,16)
  const float* qvs = (const float*)d_in[2];   // query_vox_size   (2,3)
  const float* qc  = (const float*)d_in[3];   // query_coord      (2,3,32,32,32)
  const float* mw  = (const float*)d_in[4];   // mlp_w            (2,3,176,176)
  const float* mb  = (const float*)d_in[5];   // mlp_b            (2,3,176)
  const float* pw  = (const float*)d_in[6];   // post_w           (128,45)
  const float* pb  = (const float*)d_in[7];   // post_b           (45)
  float* out = (float*)d_out;

  _Float16* wsw  = (_Float16*)d_ws;
  _Float16* wspw = wsw + WSW_HALVES;
  float*    wsmn = (float*)(wsw + WSW_HALVES + WSPW_HALVES);

  k_init_mn<<<1, 64, 0, stream>>>(wsmn);
  const int totd = N_LAYER*KK_MLP*NT_MLP*256 + KK_POST*NT_POST*256;
  k_swizzle<<<(totd + 255)/256, 256, 0, stream>>>(mw, pw, wsw, wspw);
  k_minred<<<dim3(256, 8), 256, 0, stream>>>(ext, qc, wsmn);
  k_main<<<4096, 128, 0, stream>>>(cv, ext, qvs, qc, mb, pb, wsw, wspw, wsmn, out);
}
